// _Coupling_80118319939914
// MI455X (gfx1250) — compile-verified, matrix operand order checked by reading, not run on hardware
//
#include <hip/hip_runtime.h>
#include <hip/hip_bf16.h>
#include <math.h>

typedef __bf16 bf16_t;
typedef __attribute__((ext_vector_type(16))) __bf16 v16bf;
typedef __attribute__((ext_vector_type(8)))  __bf16 bf16x8;
typedef __attribute__((ext_vector_type(2)))  __bf16 bf16x2;
typedef __attribute__((ext_vector_type(8)))  float   v8f;

#define BATCH   65536
#define DIM     256
#define CONDN   128
#define HID     512
#define NA      128
#define NB      128
#define K1      (NA + CONDN)   /* 256 */
#define N3      (2 * NB)       /* 256 */

#define TM       128           /* batch rows per workgroup        */
#define NTHREADS 256           /* 8 wave32 -> 2 waves per SIMD32  */
#define NP       64            /* weight N-panel width            */
#define NTILES   (NP / 16)     /* acc tiles per wave = 4          */

/* LDS regions (bytes) — exactly the 320 KB workgroup budget.
 * A (128KB): h0 (first 64KB), reused for h2  (layer2 reads h1 only)
 * B (128KB): h1, reused for fp32 st          (layer3 reads h2 only)
 * W ( 64KB): staged weight panel (max 512x64 bf16); reused for the
 *            logdet reduction scratch after the last GEMM barrier. */
#define OFF_A    0
#define OFF_B    (128 * 1024)
#define OFF_W    (256 * 1024)
#define LDS_BYTES (320 * 1024)

/* Build a 16x32-bf16 A/B fragment for one lane from row-major LDS.
 * Per ISA layout: lane holds two contiguous 8-element (16B) K-chunks:
 * k = kb + (lane>>4)*8 + {0..7} and the same +16. */
__device__ __forceinline__ v16bf frag_from_lds(const bf16_t* p) {
    bf16x8 lo = *(const bf16x8*)(p);
    bf16x8 hi = *(const bf16x8*)(p + 16);
    v16bf f;
#pragma unroll
    for (int i = 0; i < 8; ++i) { f[i] = lo[i]; f[i + 8] = hi[i]; }
    return f;
}

/* One fused GEMM layer: Out = act( Alds[TMxK](bf16) @ Wglob[KxN](f32) + bias ).
 * Weights staged per 64-col panel into LDS transposed as bf16 so the
 * B-fragment is two contiguous ds_load_b128 per lane. Each wave owns a
 * 16-row M-tile and 4 accumulators (64 N-columns). The k-loop is FULLY
 * unrolled (K is a compile-time constant after inlining): the straight-line
 * SSA body lets the scheduler hoist ds_loads across k-steps for wait slack
 * without any loop-carried register rotation (no v_mov walls). */
__device__ __forceinline__ void wave_gemm(
    const bf16_t* __restrict__ Alds, const float* __restrict__ Wglob,
    const float* __restrict__ bias, bf16_t* Wp,
    bf16_t* OutB, float* OutF, int K, int N, bool relu, bool out_f32)
{
    const int tid  = threadIdx.x;
    const int lane = tid & 31;
    const int wid  = tid >> 5;
    const int m0   = wid * 16;
    const int row  = lane & 15;
    const int half = lane >> 4;

    const bf16_t* arow = Alds + (m0 + row) * K + half * 8;

    for (int nb = 0; nb < N; nb += NP) {
        __syncthreads();                      /* Wp free / inputs ready */
        /* cooperative stage: Wp[c*K + k] = bf16(W[k][nb+c]), transposed,
         * two adjacent k packed per dword store. */
        const int nvec = (K * NP) / 8;        /* (k-pair, 4-col) units */
        for (int v = tid; v < nvec; v += NTHREADS) {
            const int c4 = (v & (NP / 4 - 1)) * 4;
            const int k  = (v >> 4) * 2;
            const float4 w0 = *(const float4*)(Wglob + (size_t)k * N + nb + c4);
            const float4 w1 = *(const float4*)(Wglob + (size_t)(k + 1) * N + nb + c4);
#pragma unroll
            for (int i = 0; i < 4; ++i) {
                const float a0 = (&w0.x)[i];
                const float a1 = (&w1.x)[i];
                bf16x2 t;
                t[0] = (bf16_t)a0;
                t[1] = (bf16_t)a1;
                *(bf16x2*)(Wp + (c4 + i) * K + k) = t;
            }
        }
        __syncthreads();

        v8f acc[NTILES];
        const v8f z = {0.f, 0.f, 0.f, 0.f, 0.f, 0.f, 0.f, 0.f};
#pragma unroll
        for (int nt = 0; nt < NTILES; ++nt) acc[nt] = z;

        const bf16_t* brow = Wp + row * K + half * 8;

#pragma unroll
        for (int kb = 0; kb < K; kb += 32) {
            const v16bf a = frag_from_lds(arow + kb);
#pragma unroll
            for (int nt = 0; nt < NTILES; ++nt) {
                const v16bf b = frag_from_lds(brow + nt * 16 * K + kb);
                acc[nt] = __builtin_amdgcn_wmma_f32_16x16x32_bf16(
                    false, a, false, b, (short)0, acc[nt], false, false);
            }
        }

        /* C/D layout: VGPR r -> M = m0 + r + 8*half, N = lane&15 */
#pragma unroll
        for (int nt = 0; nt < NTILES; ++nt) {
            const int col = nb + nt * 16 + row;
            const float bv = bias[col];
#pragma unroll
            for (int r = 0; r < 8; ++r) {
                const int m = m0 + r + half * 8;
                float vv = acc[nt][r] + bv;
                if (relu) vv = fmaxf(vv, 0.f);
                if (out_f32) OutF[m * N + col] = vv;
                else         OutB[m * N + col] = (bf16_t)vv;
            }
        }
    }
    __syncthreads();   /* layer output visible to next consumer */
}

__global__ void __launch_bounds__(NTHREADS)
coupling_kernel(const float* __restrict__ x, const float* __restrict__ cond,
                const float* __restrict__ W1, const float* __restrict__ b1,
                const float* __restrict__ W2, const float* __restrict__ b2,
                const float* __restrict__ W3, const float* __restrict__ b3,
                float* __restrict__ y, float* __restrict__ logdet)
{
    extern __shared__ __align__(16) char smem[];
    bf16_t* h0  = (bf16_t*)(smem + OFF_A);   /* 128 x 256 bf16 = 64KB  */
    bf16_t* h1  = (bf16_t*)(smem + OFF_B);   /* 128 x 512 bf16 = 128KB */
    bf16_t* h2  = (bf16_t*)(smem + OFF_A);   /* 128 x 512 bf16 = 128KB */
    bf16_t* Wp  = (bf16_t*)(smem + OFF_W);   /* <= 512 x 64 bf16 = 64KB */
    float*  st  = (float*)(smem + OFF_B);    /* 128 x 256 f32 = 128KB  */
    float*  red = (float*)(smem + OFF_W);    /* reuses Wp after GEMM3  */

    const int tid  = threadIdx.x;
    const int row0 = blockIdx.x * TM;

    /* pack h0 = [xa | cond] as bf16 ; xa = even columns of x */
    for (int f = tid; f < TM * K1; f += NTHREADS) {
        const int j = f & (K1 - 1);
        const int r = f >> 8;               /* f / K1, K1 == 256 */
        float v;
        if (j < NA) v = x[(size_t)(row0 + r) * DIM + 2 * j];
        else        v = cond[(size_t)(row0 + r) * CONDN + (j - NA)];
        h0[f] = (bf16_t)v;
    }
    /* producer barrier is the first __syncthreads inside wave_gemm */

    wave_gemm(h0, W1, b1, Wp, h1, nullptr, K1,  HID, true,  false);
    wave_gemm(h1, W2, b2, Wp, h2, nullptr, HID, HID, true,  false);
    wave_gemm(h2, W3, b3, Wp, nullptr, st,  HID, N3,  false, true);

    /* epilogue: s = tanh(st[:, :128]); t = st[:, 128:]; interleave outputs */
    const int r  = tid >> 1;                /* row 0..127       */
    const int c0 = (tid & 1) * 64;          /* column half      */
    float lsum = 0.f;
    const size_t gbase = (size_t)(row0 + r) * DIM;
    for (int j = c0; j < c0 + 64; ++j) {
        const float s   = tanhf(st[r * N3 + j]);
        const float tt  = st[r * N3 + NB + j];
        const float xav = x[gbase + 2 * j];
        const float xbv = x[gbase + 2 * j + 1];
        y[gbase + 2 * j]     = xav;                 /* pass-through lane */
        y[gbase + 2 * j + 1] = xbv * expf(s) + tt;  /* transformed lane  */
        lsum += s;
    }
    red[tid] = lsum;
    __syncthreads();
    if (tid < TM) logdet[row0 + tid] = red[2 * tid] + red[2 * tid + 1];
}

extern "C" void kernel_launch(void* const* d_in, const int* in_sizes, int n_in,
                              void* d_out, int out_size, void* d_ws, size_t ws_size,
                              hipStream_t stream) {
    (void)in_sizes; (void)n_in; (void)out_size; (void)d_ws; (void)ws_size;
    const float* x    = (const float*)d_in[0];
    const float* cond = (const float*)d_in[1];
    /* d_in[2] is the mask; statically even/odd, unused */
    const float* W1 = (const float*)d_in[3];
    const float* b1 = (const float*)d_in[4];
    const float* W2 = (const float*)d_in[5];
    const float* b2 = (const float*)d_in[6];
    const float* W3 = (const float*)d_in[7];
    const float* b3 = (const float*)d_in[8];

    float* y      = (float*)d_out;
    float* logdet = y + (size_t)BATCH * DIM;

    (void)hipFuncSetAttribute((const void*)coupling_kernel,
                              hipFuncAttributeMaxDynamicSharedMemorySize,
                              LDS_BYTES);

    coupling_kernel<<<dim3(BATCH / TM), dim3(NTHREADS), LDS_BYTES, stream>>>(
        x, cond, W1, b1, W2, b2, W3, b3, y, logdet);
}
